// PointEncoder_87995289960540
// MI455X (gfx1250) — compile-verified, matrix operand order checked by reading, not run on hardware
//
#include <hip/hip_runtime.h>
#include <hip/hip_bf16.h>
#include <math.h>
#include <stdint.h>

typedef _Float16 v16h __attribute__((ext_vector_type(16)));
typedef float    v8f  __attribute__((ext_vector_type(8)));

#if defined(__gfx1250__) && __has_builtin(__builtin_amdgcn_global_load_async_to_lds_b32) && \
    __has_builtin(__builtin_amdgcn_s_wait_asynccnt)
#define USE_ASYNC_LDS 1
typedef __attribute__((address_space(1))) int as1_int;
typedef __attribute__((address_space(3))) int as3_int;
#else
#define USE_ASYNC_LDS 0
#endif

// ---------------------------------------------------------------------------
// WMMA 16x16x32 f16 fragment index math (CDNA5 ISA 7.12.2, 16-bit layouts).
// A-frag: lane L holds row m = L&15 (half = L>>4 selects K sub-range).
//   element e (0..15): vgpr g=e>>1, pos=e&1;
//   k = (g<4 ? 0 : 16) + ((g&3)*2 + pos) + (half*8)
// B-frag: same k mapping; lane L holds column n = L&15.
// C/D-frag (f32): lane L holds col n = L&15; vgpr r holds row m = r + 8*(L>>4).
// ---------------------------------------------------------------------------
__device__ __forceinline__ int wmma_kmap(int e, int half) {
  int g = e >> 1, pos = e & 1;
  return ((g < 4) ? 0 : 16) + ((g & 3) * 2 + pos) + half * 8;
}

__device__ __forceinline__ v8f wmma_f16(v16h a, v16h b, v8f c) {
  return __builtin_amdgcn_wmma_f32_16x16x32_f16(false, a, false, b, (short)0, c,
                                                false, false);
}

// ---------------------------------------------------------------------------
// trans_enc: 20 rows of relu(x[12] @ te_w1 + b1) @ te_w2 + b2 -> T[20][256]
// ---------------------------------------------------------------------------
__global__ void trans_enc_kernel(const float* __restrict__ tm,
                                 const float* __restrict__ w1,
                                 const float* __restrict__ b1,
                                 const float* __restrict__ w2,
                                 const float* __restrict__ b2,
                                 float* __restrict__ T) {
  const int rowp = blockIdx.x;      // 0..19 -> (n, jidx)
  const int tid  = threadIdx.x;     // 0..255
  const int n = rowp >> 2, jidx = rowp & 3;
  const int j = (jidx < n) ? jidx : jidx + 1;
  __shared__ float hid[256];
  const float* x = tm + (n * 5 + j) * 16;   // top-3 rows of 4x4 = first 12 elems
  float h = b1[tid];
  for (int t = 0; t < 12; ++t) h += x[t] * w1[t * 256 + tid];
  hid[tid] = fmaxf(h, 0.0f);
  __syncthreads();
  float o = b2[tid];
  for (int k = 0; k < 256; ++k) o += hid[k] * w2[k * 256 + tid];
  T[rowp * 256 + tid] = o;
}

// ---------------------------------------------------------------------------
// ae hidden: H[10240][256] = relu(gather(i2j_anchor)[r][8] @ ae_w1 + b1)
// ---------------------------------------------------------------------------
__global__ void ae_hidden_kernel(const float* __restrict__ i2j,
                                 const float* __restrict__ w1,
                                 const float* __restrict__ b1,
                                 float* __restrict__ H) {
  const int idx = blockIdx.x * blockDim.x + threadIdx.x;  // < 10240*256
  const int r = idx >> 8, c = idx & 255;
  const int n = r / 2048, kk = r & 2047, jidx = kk >> 9, a = kk & 511;
  const int j = (jidx < n) ? jidx : jidx + 1;
  const float* x = i2j + (((n * 5 + j) * 512 + a) << 3);
  float h = b1[c];
  for (int t = 0; t < 8; ++t) h += x[t] * w1[t * 256 + c];
  H[idx] = fmaxf(h, 0.0f);
}

// ---------------------------------------------------------------------------
// WMMA GEMM: C[M][256] = X[M][256] @ W[256][256] + bias (+extra)
// One wave per 16x64 output tile: A-frag reused across 4 B-frags per K-step.
//   mode 0: + bias only
//   mode 1: + bias + trans_enc[(row/2048)*4 + (row%2048)/512][col]
//   mode 2: + bias + extra[row*256 + col]   (elementwise residual)
// grid = (M/16, 4), block = 32
// ---------------------------------------------------------------------------
__global__ void gemm256_wmma(const float* __restrict__ X,
                             const float* __restrict__ W,
                             const float* __restrict__ bias,
                             const float* __restrict__ extra,
                             float* __restrict__ C, int M, int mode) {
  const int lane = threadIdx.x;
  const int half = lane >> 4, l16 = lane & 15;
  const int row0 = blockIdx.x << 4;
  const int col0 = blockIdx.y << 6;   // 64 cols per wave

  v8f acc0{}, acc1{}, acc2{}, acc3{};
  const int arow = row0 + l16;

  for (int kt = 0; kt < 8; ++kt) {
    if (kt + 1 < 8) {
      // pull next K-slab into the near (WGP) cache while this one computes
      __builtin_prefetch(X + arow * 256 + (kt + 1) * 32, 0, 3);
      __builtin_prefetch(W + ((kt + 1) * 32 + lane) * 256 + col0, 0, 3);
    }
    v16h a, b0, b1, b2, b3;
#pragma unroll
    for (int e = 0; e < 16; ++e) {
      const int k = kt * 32 + wmma_kmap(e, half);
      a[e] = (_Float16)X[arow * 256 + k];
      const float* wr = W + k * 256 + col0 + l16;
      b0[e] = (_Float16)wr[0];
      b1[e] = (_Float16)wr[16];
      b2[e] = (_Float16)wr[32];
      b3[e] = (_Float16)wr[48];
    }
    acc0 = wmma_f16(a, b0, acc0);
    acc1 = wmma_f16(a, b1, acc1);
    acc2 = wmma_f16(a, b2, acc2);
    acc3 = wmma_f16(a, b3, acc3);
  }

#pragma unroll
  for (int t = 0; t < 4; ++t) {
    const v8f at = (t == 0) ? acc0 : (t == 1) ? acc1 : (t == 2) ? acc2 : acc3;
#pragma unroll
    for (int r = 0; r < 8; ++r) {
      const int row = row0 + r + half * 8;
      const int col = col0 + t * 16 + l16;
      float v = at[r] + bias[col];
      if (mode == 1) {
        const int n = row / 2048, jidx = (row & 2047) >> 9;
        v += extra[(n * 4 + jidx) * 256 + col];
      } else if (mode == 2) {
        v += extra[row * 256 + col];
      }
      C[row * 256 + col] = v;
    }
  }
}

// ---------------------------------------------------------------------------
// Gather neighbor features: vin = gather(b_feature); kin += gather(b_feature)
// ---------------------------------------------------------------------------
__global__ void gather_kv_kernel(const float* __restrict__ bf,
                                 float* __restrict__ kin,
                                 float* __restrict__ vin) {
  const int idx = blockIdx.x * blockDim.x + threadIdx.x;  // < 10240*256
  const int r = idx >> 8, c = idx & 255;
  const int n = r / 2048, kk = r & 2047, jidx = kk >> 9, a = kk & 511;
  const int j = (jidx < n) ? jidx : jidx + 1;
  const float g = bf[(((j << 9) + a) << 8) + c];
  vin[idx] = g;
  kin[idx] += g;
}

__global__ void qin_kernel(const float* __restrict__ bf,
                           const float* __restrict__ ae,
                           float* __restrict__ qin) {
  const int idx = blockIdx.x * blockDim.x + threadIdx.x;  // < 2560*256
  qin[idx] = bf[idx] + ae[idx];
}

// ---------------------------------------------------------------------------
// Flash attention: one 1024-thread block per (n, h); wave w = q-tile w.
// K/V tiles staged into LDS once per block via async global->LDS DMA,
// DOUBLE-BUFFERED: tile t+1's DMA is issued before computing on tile t,
// so the ASYNCcnt wait at the end of the iteration overlaps DMA with
// 4 WMMAs + softmax. One barrier protects both buffer handoffs.
// ---------------------------------------------------------------------------
__global__ void __launch_bounds__(1024)
attn_kernel(const float* __restrict__ q_all,
            const float* __restrict__ k_all,
            const float* __restrict__ v_all,
            float* __restrict__ o_all) {
  const int h = blockIdx.x & 7, n = blockIdx.x >> 3;   // 40 blocks
  const int tid  = threadIdx.x;
  const int w    = tid >> 5;          // wave id == q-tile id (0..31)
  const int lane = tid & 31;
  const int half = lane >> 4, l16 = lane & 15;
  const float scale = 0.17677669529663687f;  // 1/sqrt(32)

  __shared__ float    Kt[2][32][33];          // ping-pong [key][feat], padded
  __shared__ float    Vt[2][32][33];
  __shared__ _Float16 Pl[32][16][34];         // per-wave probs [qrow][key]

  const int skey  = tid >> 5;                 // staging coords: 1 float/thread
  const int sfeat = tid & 31;

  // stage one 32-key tile of K and V into LDS buffer `buf`
  auto stage = [&](int kt, int buf) {
    const int krow = n * 2048 + kt * 32 + skey;
    const float* kg = k_all + krow * 256 + h * 32 + sfeat;
    const float* vg = v_all + krow * 256 + h * 32 + sfeat;
#if USE_ASYNC_LDS
    __builtin_amdgcn_global_load_async_to_lds_b32(
        (as1_int*)(uintptr_t)kg,
        (as3_int*)(uint32_t)(uintptr_t)&Kt[buf][skey][sfeat], 0, 0);
    __builtin_amdgcn_global_load_async_to_lds_b32(
        (as1_int*)(uintptr_t)vg,
        (as3_int*)(uint32_t)(uintptr_t)&Vt[buf][skey][sfeat], 0, 0);
#else
    Kt[buf][skey][sfeat] = *kg;
    Vt[buf][skey][sfeat] = *vg;
#endif
  };

  // Q A-fragment (16 rows x K=32 features), loaded once per wave
  v16h qf;
  {
    const int row = n * 512 + w * 16 + l16;
#pragma unroll
    for (int e = 0; e < 16; ++e)
      qf[e] = (_Float16)q_all[row * 256 + h * 32 + wmma_kmap(e, half)];
  }

  float m_i[8], l_i[8];
  v8f o0{}, o1{};
  for (int r = 0; r < 8; ++r) { m_i[r] = -INFINITY; l_i[r] = 0.0f; }

  // prologue: stage tile 0
  stage(0, 0);
#if USE_ASYNC_LDS
  __builtin_amdgcn_s_wait_asynccnt(0);
#endif
  __syncthreads();

  for (int kt = 0; kt < 64; ++kt) {     // 2048 keys in tiles of 32
    const int cur = kt & 1;
    if (kt + 1 < 64) stage(kt + 1, cur ^ 1);   // fire-and-forget DMA

    // S = Q K^T : two 16x16 tiles; B-frag column = key index (from LDS)
    v16h kb0, kb1;
#pragma unroll
    for (int e = 0; e < 16; ++e) {
      const int k = wmma_kmap(e, half);
      kb0[e] = (_Float16)Kt[cur][l16][k];
      kb1[e] = (_Float16)Kt[cur][l16 + 16][k];
    }
    v8f s0{}, s1{};
    s0 = wmma_f16(qf, kb0, s0);
    s1 = wmma_f16(qf, kb1, s1);

    // online softmax; each lane owns rows r+8*half, column l16 of S tiles
#pragma unroll
    for (int r = 0; r < 8; ++r) {
      const float a0 = s0[r] * scale, a1 = s1[r] * scale;
      float vmax = fmaxf(a0, a1);
      for (int d = 1; d < 16; d <<= 1) vmax = fmaxf(vmax, __shfl_xor(vmax, d, 32));
      const float mn = fmaxf(m_i[r], vmax);
      const float p0 = __expf(a0 - mn), p1 = __expf(a1 - mn);
      float ps = p0 + p1;
      for (int d = 1; d < 16; d <<= 1) ps += __shfl_xor(ps, d, 32);
      const float corr = __expf(m_i[r] - mn);
      l_i[r] = l_i[r] * corr + ps;
      o0[r] *= corr;
      o1[r] *= corr;
      m_i[r] = mn;
      const int m = r + half * 8;
      Pl[w][m][l16]      = (_Float16)p0;   // keys kt*32 + 0..15
      Pl[w][m][l16 + 16] = (_Float16)p1;   // keys kt*32 + 16..31
    }
    __syncthreads();                    // P restripe C-layout -> A-layout

    // O += P V : P A-frag from LDS; V B-frags from LDS
    v16h pf, vb0, vb1;
#pragma unroll
    for (int e = 0; e < 16; ++e) {
      const int k = wmma_kmap(e, half);       // key within tile
      pf[e]  = Pl[w][l16][k];
      vb0[e] = (_Float16)Vt[cur][k][l16];
      vb1[e] = (_Float16)Vt[cur][k][16 + l16];
    }
    o0 = wmma_f16(pf, vb0, o0);
    o1 = wmma_f16(pf, vb1, o1);

    // end of iteration: next tile's DMA done + everyone finished `cur`
#if USE_ASYNC_LDS
    __builtin_amdgcn_s_wait_asynccnt(0);
#endif
    __syncthreads();
  }

#pragma unroll
  for (int r = 0; r < 8; ++r) {
    const int row = n * 512 + w * 16 + r + half * 8;
    const float inv = 1.0f / l_i[r];
    o_all[row * 256 + h * 32 + l16]      = o0[r] * inv;
    o_all[row * 256 + h * 32 + 16 + l16] = o1[r] * inv;
  }
}

// ---------------------------------------------------------------------------
// Heads: per anchor row, 11 narrow dots + box construction.
// out layout: cls@0, box@2560, dirp@20480, output@25600, res@46080
// ---------------------------------------------------------------------------
__global__ void heads_kernel(const float* __restrict__ res,
                             const float* __restrict__ b_anchor,
                             const float* __restrict__ w_reg,
                             const float* __restrict__ b_reg,
                             const float* __restrict__ w_cls,
                             const float* __restrict__ b_cls,
                             const float* __restrict__ w_dir,
                             const float* __restrict__ b_dir,
                             float* __restrict__ out) {
  const int row = blockIdx.x;          // < 2560
  const int t = threadIdx.x;           // 0..31
  __shared__ float o8[8];
  const float* r_ = res + row * 256;
  if (t < 8) {
    float acc = b_reg[t];
    for (int k = 0; k < 256; ++k) acc += r_[k] * w_reg[k * 8 + t];
    const float ov = acc + b_anchor[row * 8 + t];
    out[25600 + row * 8 + t] = ov;     // output
    o8[t] = ov;
  } else if (t == 8) {
    float acc = b_cls[0];
    for (int k = 0; k < 256; ++k) acc += r_[k] * w_cls[k];
    out[row] = acc;                    // cls
  } else if (t < 11) {
    const int c = t - 9;
    float acc = b_dir[c];
    for (int k = 0; k < 256; ++k) acc += r_[k] * w_dir[k * 2 + c];
    out[20480 + row * 2 + c] = acc;    // dirp
  }
  __syncthreads();
  if (t < 7) {
    float v;
    if (t < 3)      v = o8[t];
    else if (t < 6) v = expf(o8[t]);
    else            v = atan2f(o8[6], o8[7]);  // SIN_YAW=6, COS_YAW=7
    out[2560 + row * 7 + t] = v;       // box
  }
}

// ---------------------------------------------------------------------------
extern "C" void kernel_launch(void* const* d_in, const int* in_sizes, int n_in,
                              void* d_out, int out_size, void* d_ws, size_t ws_size,
                              hipStream_t stream) {
  const float* i2j_anchor   = (const float*)d_in[0];
  const float* b_anchor     = (const float*)d_in[1];
  const float* b_feature    = (const float*)d_in[2];
  const float* transform    = (const float*)d_in[3];
  // d_in[4] lidar_range: unused by reference math
  const float* anchor_embed = (const float*)d_in[5];
  const float* te_w1 = (const float*)d_in[6];
  const float* te_b1 = (const float*)d_in[7];
  const float* te_w2 = (const float*)d_in[8];
  const float* te_b2 = (const float*)d_in[9];
  const float* ae_w1 = (const float*)d_in[10];
  const float* ae_b1 = (const float*)d_in[11];
  const float* ae_w2 = (const float*)d_in[12];
  const float* ae_b2 = (const float*)d_in[13];
  const float* wq = (const float*)d_in[14];
  const float* bq = (const float*)d_in[15];
  const float* wk = (const float*)d_in[16];
  const float* bk = (const float*)d_in[17];
  const float* wv = (const float*)d_in[18];
  const float* bv = (const float*)d_in[19];
  const float* wo = (const float*)d_in[20];
  const float* bo = (const float*)d_in[21];
  const float* w_reg = (const float*)d_in[22];
  const float* b_reg = (const float*)d_in[23];
  const float* w_cls = (const float*)d_in[24];
  const float* b_cls = (const float*)d_in[25];
  const float* w_dir = (const float*)d_in[26];
  const float* b_dir = (const float*)d_in[27];

  float* out = (float*)d_out;
  float* ws  = (float*)d_ws;

  // workspace layout (floats)
  float* T   = ws;                    //    20*256 : trans_enc
  float* B0  = ws + 5120;             // 10240*256 : ae hidden -> later k_all
  float* B1  = ws + 2626560;          // 10240*256 : key_pos -> kin (in place)
  float* B2  = ws + 5248000;          // 10240*256 : vin (gathered key_feat)
  float* B3  = ws + 7869440;          // 10240*256 : v_all
  float* B4  = ws + 10490880;         //  2560*256 : qin
  float* B5  = ws + 11146240;         //  2560*256 : q_all
  float* B6  = ws + 11801600;         //  2560*256 : ctx (attention output)
  float* res = out + 46080;           //  2560*256 : residual output slot

  trans_enc_kernel<<<20, 256, 0, stream>>>(transform, te_w1, te_b1, te_w2, te_b2, T);
  ae_hidden_kernel<<<10240, 256, 0, stream>>>(i2j_anchor, ae_w1, ae_b1, B0);
  // key_pos = hidden @ ae_w2 + ae_b2 + trans_enc(broadcast over anchors)
  gemm256_wmma<<<dim3(640, 4), 32, 0, stream>>>(B0, ae_w2, ae_b2, T, B1, 10240, 1);
  // kin = key_feat + key_pos (in place into B1); vin = key_feat
  gather_kv_kernel<<<10240, 256, 0, stream>>>(b_feature, B1, B2);
  qin_kernel<<<2560, 256, 0, stream>>>(b_feature, anchor_embed, B4);
  // projections
  gemm256_wmma<<<dim3(160, 4), 32, 0, stream>>>(B4, wq, bq, nullptr, B5, 2560, 0);
  gemm256_wmma<<<dim3(640, 4), 32, 0, stream>>>(B1, wk, bk, nullptr, B0, 10240, 0);
  gemm256_wmma<<<dim3(640, 4), 32, 0, stream>>>(B2, wv, bv, nullptr, B3, 10240, 0);
  // attention: q=B5, k=B0, v=B3 -> ctx=B6
  attn_kernel<<<40, 1024, 0, stream>>>(B5, B0, B3, B6);
  // res = b_feature + ctx @ wo + bo  (written directly into d_out slot)
  gemm256_wmma<<<dim3(160, 4), 32, 0, stream>>>(B6, wo, bo, b_feature, res, 2560, 2);
  // heads: output/cls/dirp/box
  heads_kernel<<<2560, 32, 0, stream>>>(res, b_anchor, w_reg, b_reg, w_cls, b_cls,
                                        w_dir, b_dir, out);
}
